// DenseStructuralGATLayer_50835232915949
// MI455X (gfx1250) — matrix operand order repaired and checked by reading, not compiled, not run on hardware
//
#include <hip/hip_runtime.h>
#include <hip/hip_bf16.h>

typedef __attribute__((ext_vector_type(16))) _Float16 v16h;
typedef __attribute__((ext_vector_type(8)))  float    v8f;

#define NND   8192
#define FIN   512
#define FOUT  256
#define NEG_SLOPE 0.2f

// pack 8 floats (two float4) into halves [base..base+7] of a v16h
__device__ __forceinline__ void pack8(v16h& v, int base, float4 lo, float4 hi) {
    v[base+0] = (_Float16)lo.x; v[base+1] = (_Float16)lo.y;
    v[base+2] = (_Float16)lo.z; v[base+3] = (_Float16)lo.w;
    v[base+4] = (_Float16)hi.x; v[base+5] = (_Float16)hi.y;
    v[base+6] = (_Float16)hi.z; v[base+7] = (_Float16)hi.w;
}

// async global -> LDS copy of 16 bytes (CDNA5, tracked by ASYNCcnt)
__device__ __forceinline__ void async_copy_b128(unsigned lds_off, const void* gaddr) {
    asm volatile("global_load_async_to_lds_b128 %0, %1, off"
                 :: "v"(lds_off), "v"(gaddr) : "memory");
}
__device__ __forceinline__ void wait_asynccnt0() {
    asm volatile("s_wait_asynccnt 0x0" ::: "memory");
}

// ---------------- Kernel 1: H = X @ W^T  (WMMA f16, f32 accum) ----------------
// wave computes a 16x64 tile of H. 2048 waves total.
__global__ __launch_bounds__(256) void gemm_h_kernel(const float* __restrict__ X,
                                                     const float* __restrict__ W,
                                                     float* __restrict__ H) {
    const int tid  = threadIdx.x;
    const int lane = tid & 31;
    const int w    = blockIdx.x * 8 + (tid >> 5);   // global wave id, 0..2047
    const int row_tile = w >> 2;                    // 0..511
    const int col_base = (w & 3) * 64;
    const int m  = lane & 15;
    const int hs = lane >> 4;                       // 0 or 1 (lane half)
    const int row = row_tile * 16 + m;

    v8f acc[4] = {};
    for (int k = 0; k < FIN; k += 32) {
        // A: X[row][k..k+31] per ISA 16-bit A layout
        const float* xp = X + (size_t)row * FIN + k;
        float4 a0 = *(const float4*)(xp + hs * 8);
        float4 a1 = *(const float4*)(xp + hs * 8 + 4);
        float4 a2 = *(const float4*)(xp + 16 + hs * 8);
        float4 a3 = *(const float4*)(xp + 16 + hs * 8 + 4);
        v16h A;
        pack8(A, 0, a0, a1);
        pack8(A, 8, a2, a3);
        #pragma unroll
        for (int t = 0; t < 4; ++t) {
            // B: lane n holds W row (col_base + t*16 + n), K in A-layout order
            const int wrow = col_base + t * 16 + m;
            const float* wp = W + (size_t)wrow * FIN + k;
            float4 b0 = *(const float4*)(wp + hs * 8);
            float4 b1 = *(const float4*)(wp + hs * 8 + 4);
            float4 b2 = *(const float4*)(wp + 16 + hs * 8);
            float4 b3 = *(const float4*)(wp + 16 + hs * 8 + 4);
            v16h B;
            pack8(B, 0, b0, b1);
            pack8(B, 8, b2, b3);
            acc[t] = __builtin_amdgcn_wmma_f32_16x16x32_f16(
                false, A, false, B, (short)0, acc[t], false, false);
        }
    }
    #pragma unroll
    for (int t = 0; t < 4; ++t) {
        #pragma unroll
        for (int f = 0; f < 8; ++f) {
            const int r = row_tile * 16 + f + hs * 8;
            const int c = col_base + t * 16 + m;
            H[(size_t)r * FOUT + c] = acc[t][f];
        }
    }
}

// ---------------- Kernel 2a: s = H a_l, t = H a_r (one wave per row) ----------
__global__ __launch_bounds__(256) void attn_st_kernel(const float* __restrict__ H,
                                                      const float* __restrict__ aw,
                                                      float* __restrict__ s,
                                                      float* __restrict__ t) {
    const int lane = threadIdx.x & 31;
    const int row  = blockIdx.x * 8 + (threadIdx.x >> 5);
    const float* hp = H + (size_t)row * FOUT + lane * 8;
    float4 h0 = *(const float4*)(hp);
    float4 h1 = *(const float4*)(hp + 4);
    const float* al = aw + lane * 8;
    const float* ar = aw + FOUT + lane * 8;
    float4 l0 = *(const float4*)(al), l1 = *(const float4*)(al + 4);
    float4 r0 = *(const float4*)(ar), r1 = *(const float4*)(ar + 4);
    float ss = h0.x*l0.x + h0.y*l0.y + h0.z*l0.z + h0.w*l0.w
             + h1.x*l1.x + h1.y*l1.y + h1.z*l1.z + h1.w*l1.w;
    float tt = h0.x*r0.x + h0.y*r0.y + h0.z*r0.z + h0.w*r0.w
             + h1.x*r1.x + h1.y*r1.y + h1.z*r1.z + h1.w*r1.w;
    #pragma unroll
    for (int off = 16; off > 0; off >>= 1) {
        ss += __shfl_xor(ss, off);
        tt += __shfl_xor(tt, off);
    }
    if (lane == 0) { s[row] = ss; t[row] = tt; }
}

// ---------------- Kernel 2b: Ht[o][i] = (f16) H[i][o] (tiled transpose) ------
__global__ __launch_bounds__(256) void transpose_h_kernel(const float* __restrict__ H,
                                                          _Float16* __restrict__ Ht) {
    __shared__ _Float16 tile[32][33];
    const int tx = threadIdx.x & 31;
    const int ty = threadIdx.x >> 5;              // 0..7
    const int r0 = blockIdx.x * 32;               // node base
    const int c0 = blockIdx.y * 32;               // feature base
    #pragma unroll
    for (int p = 0; p < 32; p += 8)
        tile[ty + p][tx] = (_Float16)H[(size_t)(r0 + ty + p) * FOUT + c0 + tx];
    __syncthreads();
    #pragma unroll
    for (int p = 0; p < 32; p += 8)
        Ht[(size_t)(c0 + ty + p) * NND + r0 + tx] = tile[tx][ty + p];
}

// ---------------- Kernel 3: fused online-softmax + P@H (flash style) ---------
// 256 threads = 8 waves; wave w owns rows [i0, i0+16) and ALL 256 output cols
// (16 WMMA accumulators). Per 32-column step the workgroup double-buffers the
// 32x256 f16 chunk of H in LDS via CDNA5 async global->LDS loads, so staging of
// chunk j+32 overlaps compute on chunk j with a single barrier per step.
__global__ __launch_bounds__(256) void gat_attn_kernel(const float* __restrict__ M,
                                                       const float* __restrict__ s,
                                                       const float* __restrict__ tv,
                                                       const float* __restrict__ H,
                                                       const _Float16* __restrict__ Ht,
                                                       float* __restrict__ Z) {
    __shared__ _Float16 ldsB[2][256 * 32];        // 2 x 16 KB double buffer

    const int tid  = threadIdx.x;
    const int lane = tid & 31;
    const int wave = tid >> 5;
    const int m    = lane & 15;
    const int hs   = lane >> 4;
    const int i0   = blockIdx.x * 128 + wave * 16;

    const float si = s[i0 + m];                   // lane's row attention term
    float run_m = -INFINITY;
    float run_l = 0.f;
    v8f acc[16] = {};

    const int g0 = hs * 16;                       // contiguous K offsets (k = 16h + s)
    const int g1 = hs * 16 + 8;

    // cooperative async stage of Ht[0..255][j..j+31] into buffer `buf`
    auto stage = [&](int buf, int j) {
        #pragma unroll
        for (int r = 0; r < 4; ++r) {
            const int q = tid + 256 * r;
            const int o = q >> 2, part = q & 3;
            const unsigned loff =
                (unsigned)(uintptr_t)(&ldsB[buf][o * 32 + part * 8]);
            async_copy_b128(loff, (const void*)(Ht + (size_t)o * NND + j + part * 8));
        }
    };

    stage(0, 0);
    wait_asynccnt0();
    __syncthreads();

    for (int j = 0; j < NND; j += 32) {
        const int cur = (j >> 5) & 1;
        if (j + 32 < NND) stage(cur ^ 1, j + 32);   // overlap with compute

        // M tile + t chunk, contiguous K per lane half (k = 16h + s)
        const float* mp = M + (size_t)(i0 + m) * NND + j;
        __builtin_prefetch(mp + 512, 0, 0);         // stream M ~2KB ahead
        float4 M0 = *(const float4*)(mp + g0),     M1 = *(const float4*)(mp + g0 + 4);
        float4 M2 = *(const float4*)(mp + g1),     M3 = *(const float4*)(mp + g1 + 4);
        float4 T0 = *(const float4*)(tv + j + g0), T1 = *(const float4*)(tv + j + g0 + 4);
        float4 T2 = *(const float4*)(tv + j + g1), T3 = *(const float4*)(tv + j + g1 + 4);
        const float mv[16] = {M0.x,M0.y,M0.z,M0.w, M1.x,M1.y,M1.z,M1.w,
                              M2.x,M2.y,M2.z,M2.w, M3.x,M3.y,M3.z,M3.w};
        const float te[16] = {T0.x,T0.y,T0.z,T0.w, T1.x,T1.y,T1.z,T1.w,
                              T2.x,T2.y,T2.z,T2.w, T3.x,T3.y,T3.z,T3.w};

        float lg[16];
        float tmax = -INFINITY;
        #pragma unroll
        for (int e = 0; e < 16; ++e) {
            const float x  = si + te[e];
            const float lr = x > 0.f ? x : NEG_SLOPE * x;
            const float l  = mv[e] > 0.f ? mv[e] * lr : -INFINITY;
            lg[e] = l;
            tmax  = fmaxf(tmax, l);
        }
        tmax = fmaxf(tmax, __shfl_xor(tmax, 16)); // combine row's two lane-halves

        const float m_new = fmaxf(run_m, tmax);
        const float scale = (m_new == -INFINITY) ? 1.f : __expf(run_m - m_new);

        float psum = 0.f;
        v16h A;
        #pragma unroll
        for (int e = 0; e < 16; ++e) {
            const float p = (lg[e] == -INFINITY) ? 0.f : __expf(lg[e] - m_new);
            psum += p;
            A[e] = (_Float16)p;
        }
        psum += __shfl_xor(psum, 16);
        run_l = run_l * scale + psum;
        run_m = m_new;

        // rescale accumulators only when some row's max actually moved
        if (__any(scale != 1.0f)) {
            float sc[8];
            #pragma unroll
            for (int f = 0; f < 8; ++f) sc[f] = __shfl(scale, f + hs * 8);
            #pragma unroll
            for (int t2 = 0; t2 < 16; ++t2) {
                #pragma unroll
                for (int f = 0; f < 8; ++f) acc[t2][f] *= sc[f];
            }
        }

        // 16 WMMAs: acc[t2] += P(16x32) @ H_chunk(32x16) for column tile t2
        #pragma unroll
        for (int t2 = 0; t2 < 16; ++t2) {
            const v16h B = *(const v16h*)(&ldsB[cur][(t2 * 16 + m) * 32 + hs * 16]);
            acc[t2] = __builtin_amdgcn_wmma_f32_16x16x32_f16(
                false, A, false, B, (short)0, acc[t2], false, false);
        }

        wait_asynccnt0();                          // own async stores to LDS done
        __syncthreads();                           // whole tile visible to all waves
    }

    // finalize: divide by softmax denom, no-neighbor rows fall back to H[i,:]
    float lrow[8], mrow[8];
    #pragma unroll
    for (int f = 0; f < 8; ++f) {
        lrow[f] = __shfl(run_l, f + hs * 8);
        mrow[f] = __shfl(run_m, f + hs * 8);
    }
    #pragma unroll
    for (int t2 = 0; t2 < 16; ++t2) {
        const int col = t2 * 16 + m;
        #pragma unroll
        for (int f = 0; f < 8; ++f) {
            const int i = i0 + f + hs * 8;
            float v;
            if (mrow[f] == -INFINITY)
                v = H[(size_t)i * FOUT + col];     // alpha = identity row
            else
                v = acc[t2][f] / lrow[f];
            Z[(size_t)i * FOUT + col] = 1.f / (1.f + __expf(-v));
        }
    }
}

// ---------------- launch ------------------------------------------------------
extern "C" void kernel_launch(void* const* d_in, const int* in_sizes, int n_in,
                              void* d_out, int out_size, void* d_ws, size_t ws_size,
                              hipStream_t stream) {
    const float* X  = (const float*)d_in[0];   // 8192 x 512
    const float* M  = (const float*)d_in[1];   // 8192 x 8192
    const float* W  = (const float*)d_in[2];   // 256 x 512
    const float* aw = (const float*)d_in[3];   // 1 x 512
    float* Z = (float*)d_out;                  // 8192 x 256

    // workspace layout: H (f32, 8MB) | Ht (f16, 4MB) | s | t
    float*    H  = (float*)d_ws;
    _Float16* Ht = (_Float16*)(H + (size_t)NND * FOUT);
    float*    s  = (float*)(Ht + (size_t)FOUT * NND);
    float*    t  = s + NND;

    gemm_h_kernel<<<256, 256, 0, stream>>>(X, W, H);
    attn_st_kernel<<<NND / 8, 256, 0, stream>>>(H, aw, s, t);
    transpose_h_kernel<<<dim3(NND / 32, FOUT / 32), 256, 0, stream>>>(H, Ht);
    gat_attn_kernel<<<NND / 128, 256, 0, stream>>>(M, s, t, H, Ht, Z);
}
